// HTMLayer_72155450573342
// MI455X (gfx1250) — 8-bit matrix operand order repaired and checked by reading, compiled, not run on hardware
//
#include <hip/hip_runtime.h>

#define C_          1024
#define INP_        1024
#define KCELLS_     4
#define N_          4096
#define BSTEPS_     16
#define NUM_ACTIVE_ 20
#define PERM_CONN_  0.5f
#define PERM_INC_   0.1f
#define PERM_DEC_   0.1f
#define BOOST_STR_  3.0f
#define SPARSITY_   0.02f
#define ACT_TH_I_   13
#define TM_INC_     0.1f
#define TM_DECX_    0.01f

typedef __attribute__((ext_vector_type(8))) int v8i;

__global__ void k_init_seg(const float* __restrict__ seg_in,
                           float* __restrict__ seg_f,
                           unsigned char* __restrict__ seg_u8) {
  unsigned idx = blockIdx.x * blockDim.x + threadIdx.x;
  float4 v = ((const float4*)seg_in)[idx];
  ((float4*)seg_f)[idx] = v;
  uchar4 u;
  u.x = (v.x >= PERM_CONN_) ? 1 : 0;
  u.y = (v.y >= PERM_CONN_) ? 1 : 0;
  u.z = (v.z >= PERM_CONN_) ? 1 : 0;
  u.w = (v.w >= PERM_CONN_) ? 1 : 0;
  ((uchar4*)seg_u8)[idx] = u;
}

__global__ void k_init_small(const float* __restrict__ perm_in, float* __restrict__ perm,
                             const float* __restrict__ boost_in, float* __restrict__ boost,
                             const float* __restrict__ duty_in,  float* __restrict__ duty,
                             float* __restrict__ act0, float* __restrict__ act1,
                             float* __restrict__ pred0, float* __restrict__ pred1) {
  unsigned gt = blockIdx.x * blockDim.x + threadIdx.x;
  ((float4*)perm)[gt] = ((const float4*)perm_in)[gt];
  if (gt < C_) { boost[gt] = boost_in[gt]; duty[gt] = duty_in[gt]; }
  if (gt < N_) { act0[gt] = 0.f; act1[gt] = 0.f; pred0[gt] = 0.f; pred1[gt] = 0.f; }
}

__global__ void k_overlap(const float* __restrict__ perm, const float* __restrict__ pm,
                          const float* __restrict__ x, const float* __restrict__ boost,
                          float* __restrict__ overlap) {
  __shared__ float red[256];
  const int c = blockIdx.x;
  const float* prow = perm + (size_t)c * INP_;
  const float* mrow = pm   + (size_t)c * INP_;
  float s = 0.f;
  for (int i = threadIdx.x; i < INP_; i += 256) {
    float conn = (prow[i] >= PERM_CONN_) ? mrow[i] : 0.f;
    s += conn * x[i];
  }
  red[threadIdx.x] = s;
  __syncthreads();
  for (int st = 128; st > 0; st >>= 1) {
    if (threadIdx.x < st) red[threadIdx.x] += red[threadIdx.x + st];
    __syncthreads();
  }
  if (threadIdx.x == 0) overlap[c] = red[0] * boost[c];
}

__global__ void __launch_bounds__(1024)
k_topk_tm(const float* __restrict__ overlap, const float* __restrict__ pred_prev,
          float* __restrict__ active_cols_ws, int* __restrict__ active_idx,
          float* __restrict__ duty, float* __restrict__ boost,
          float* __restrict__ act_next, unsigned char* __restrict__ newact_u8,
          float* __restrict__ out_ac, float* __restrict__ out_cells,
          float* __restrict__ out_anom, float alpha) {
  __shared__ float vals[C_];
  __shared__ float rv[C_];
  __shared__ int   ri[C_];
  __shared__ float s_active[C_];
  __shared__ int   sel[NUM_ACTIVE_];
  __shared__ float s_npred;
  const int t = threadIdx.x;

  vals[t] = overlap[t];
  s_active[t] = 0.f;
  __syncthreads();

  for (int k = 0; k < NUM_ACTIVE_; ++k) {
    rv[t] = vals[t]; ri[t] = t;
    __syncthreads();
    for (int s = C_ / 2; s > 0; s >>= 1) {
      if (t < s) {
        float v2 = rv[t + s]; int i2 = ri[t + s];
        if (v2 > rv[t] || (v2 == rv[t] && i2 < ri[t])) { rv[t] = v2; ri[t] = i2; }
      }
      __syncthreads();
    }
    if (t == 0) { int w = ri[0]; sel[k] = w; vals[w] = -3.0e38f; s_active[w] = 1.f; }
    __syncthreads();
  }
  if (t < NUM_ACTIVE_) active_idx[t] = sel[t];

  float ac = s_active[t];
  active_cols_ws[t] = ac;
  out_ac[t] = ac;
  float d = duty[t];
  d = (1.f - alpha) * d + alpha * ac;
  duty[t] = d;
  boost[t] = expf(BOOST_STR_ * (SPARSITY_ - d));

  float p0 = pred_prev[KCELLS_ * t + 0];
  float p1 = pred_prev[KCELLS_ * t + 1];
  float p2 = pred_prev[KCELLS_ * t + 2];
  float p3 = pred_prev[KCELLS_ * t + 3];
  float chp = (p0 > 0.f || p1 > 0.f || p2 > 0.f || p3 > 0.f) ? 1.f : 0.f;
  float na[KCELLS_];
  na[0] = (ac > 0.f) ? ((chp > 0.f) ? p0 : 1.f) : 0.f;
  na[1] = (ac > 0.f) ? ((chp > 0.f) ? p1 : 1.f) : 0.f;
  na[2] = (ac > 0.f) ? ((chp > 0.f) ? p2 : 1.f) : 0.f;
  na[3] = (ac > 0.f) ? ((chp > 0.f) ? p3 : 1.f) : 0.f;
  #pragma unroll
  for (int k = 0; k < KCELLS_; ++k) {
    act_next[KCELLS_ * t + k]  = na[k];
    out_cells[KCELLS_ * t + k] = na[k];
    newact_u8[KCELLS_ * t + k] = (unsigned char)(na[k] > 0.f ? 1 : 0);
  }

  rv[t] = ac * chp;
  __syncthreads();
  for (int s = C_ / 2; s > 0; s >>= 1) {
    if (t < s) rv[t] += rv[t + s];
    __syncthreads();
  }
  if (t == 0) s_npred = rv[0];
  __syncthreads();
  rv[t] = ac;
  __syncthreads();
  for (int s = C_ / 2; s > 0; s >>= 1) {
    if (t < s) rv[t] += rv[t + s];
    __syncthreads();
  }
  if (t == 0) {
    float nact = rv[0];
    out_anom[0] = 1.f - s_npred / fmaxf(nact, 1.f);
  }
}

__global__ void k_perm_update(float* __restrict__ perm, const int* __restrict__ active_idx,
                              const float* __restrict__ x, const float* __restrict__ pm) {
  const int c = active_idx[blockIdx.x];
  const float* mrow = pm + (size_t)c * INP_;
  float* prow = perm + (size_t)c * INP_;
  for (int i = threadIdx.x; i < INP_; i += blockDim.x) {
    float xv = x[i];
    float m  = mrow[i];
    float p  = prow[i] + PERM_INC_ * m * xv - PERM_DEC_ * m * (1.f - xv);
    prow[i] = fminf(fmaxf(p, 0.f), 1.f);
  }
}

__global__ void __launch_bounds__(256)
k_presyn(const unsigned char* __restrict__ seg_u8,
         const unsigned char* __restrict__ newact_u8,
         float* __restrict__ pred_next, float* __restrict__ out_pred) {
  __shared__ __align__(16) unsigned char lds_na[N_];
  ((int4*)lds_na)[threadIdx.x] = ((const int4*)newact_u8)[threadIdx.x];
  __syncthreads();

  const int wave = (int)(blockIdx.x * (blockDim.x >> 5)) + (int)(threadIdx.x >> 5);
  const int lane = threadIdx.x & 31;
  const int half = lane >> 4;
  const int m    = lane & 15;
  const size_t rowbase = ((size_t)wave * 16 + m) * (size_t)N_ + (size_t)half * 16;

  v8i acc = {0, 0, 0, 0, 0, 0, 0, 0};
  for (int k0 = 0; k0 < N_; k0 += 64) {
    const unsigned char* ap = seg_u8 + rowbase + k0;
    __builtin_prefetch(ap + 64, 0, 1);
    unsigned long long q0 = *(const unsigned long long*)(ap + 0);
    unsigned long long q1 = *(const unsigned long long*)(ap + 8);
    unsigned long long q2 = *(const unsigned long long*)(ap + 32);
    unsigned long long q3 = *(const unsigned long long*)(ap + 40);
    v8i a, b;
    a[0] = (int)(unsigned)q0; a[1] = (int)(q0 >> 32);
    a[2] = (int)(unsigned)q1; a[3] = (int)(q1 >> 32);
    a[4] = (int)(unsigned)q2; a[5] = (int)(q2 >> 32);
    a[6] = (int)(unsigned)q3; a[7] = (int)(q3 >> 32);
    int4 b0 = *(const int4*)(lds_na + k0 + half * 16);
    int4 b1 = *(const int4*)(lds_na + k0 + 32 + half * 16);
    b[0] = b0.x; b[1] = b0.y; b[2] = b0.z; b[3] = b0.w;
    b[4] = b1.x; b[5] = b1.y; b[6] = b1.z; b[7] = b1.w;
    acc = __builtin_amdgcn_wmma_i32_16x16x64_iu8(false, a, false, b, acc,
                                                 false, false);
  }

  if (m == 0) {
    const int rbase = wave * 16 + half * 8;
    #pragma unroll
    for (int r = 0; r < 8; ++r) {
      float pred = (acc[r] >= ACT_TH_I_) ? 1.0f : 0.0f;
      pred_next[rbase + r] = pred;
      out_pred[rbase + r]  = pred;
    }
  }
}

__global__ void k_seg_update(float* __restrict__ seg_f, unsigned char* __restrict__ seg_u8,
                             const float* __restrict__ act_prev,
                             const float* __restrict__ act_next) {
  const int row = blockIdx.x;
  if (act_prev[row] <= 0.f) return;
  const size_t base = (size_t)row * N_;
  for (int j = threadIdx.x; j < N_; j += blockDim.x) {
    float na = act_next[j];
    float s  = seg_f[base + j] + TM_INC_ * na - TM_DECX_ * (1.f - na);
    s = fminf(fmaxf(s, 0.f), 1.f);
    seg_f[base + j]  = s;
    seg_u8[base + j] = (s >= PERM_CONN_) ? (unsigned char)1 : (unsigned char)0;
  }
}

extern "C" void kernel_launch(void* const* d_in, const int* in_sizes, int n_in,
                              void* d_out, int out_size, void* d_ws, size_t ws_size,
                              hipStream_t stream) {
  const float* in_sdr   = (const float*)d_in[0];
  const float* in_perm  = (const float*)d_in[1];
  const float* in_pm    = (const float*)d_in[2];
  const float* in_boost = (const float*)d_in[3];
  const float* in_duty  = (const float*)d_in[4];
  const float* in_seg   = (const float*)d_in[5];

  char* ws = (char*)d_ws;
  float*         seg_f   = (float*)(ws);
  unsigned char* seg_u8  = (unsigned char*)(ws + ((size_t)64 << 20));
  float*         perm    = (float*)(ws + ((size_t)80 << 20));
  float*         boost   = (float*)(ws + ((size_t)84 << 20));
  float*         duty    = boost + C_;
  float*         overlap = duty + C_;
  float*         acols   = overlap + C_;
  int*           aidx    = (int*)(acols + C_);
  float*         act0    = (float*)(aidx + 32);
  float*         act1    = act0 + N_;
  float*         pred0   = act1 + N_;
  float*         pred1   = pred0 + N_;
  unsigned char* na_u8   = (unsigned char*)(pred1 + N_);

  float* out_ac    = (float*)d_out;
  float* out_cells = out_ac    + BSTEPS_ * C_;
  float* out_pred  = out_cells + BSTEPS_ * N_;
  float* out_anom  = out_pred  + BSTEPS_ * N_;

  k_init_seg  <<<16384, 256, 0, stream>>>(in_seg, seg_f, seg_u8);
  k_init_small<<<1024, 256, 0, stream>>>(in_perm, perm, in_boost, boost, in_duty, duty,
                                         act0, act1, pred0, pred1);

  for (int t = 0; t < BSTEPS_; ++t) {
    const float* x = in_sdr + (size_t)t * INP_;
    float* act_prev  = (t & 1) ? act1  : act0;
    float* act_next  = (t & 1) ? act0  : act1;
    float* pred_prev = (t & 1) ? pred1 : pred0;
    float* pred_next = (t & 1) ? pred0 : pred1;
    const float it = (float)(t + 1);
    const float alpha = 1.0f / (it < 1000.0f ? it : 1000.0f);

    k_overlap    <<<C_, 256, 0, stream>>>(perm, in_pm, x, boost, overlap);
    k_topk_tm    <<<1, 1024, 0, stream>>>(overlap, pred_prev, acols, aidx, duty, boost,
                                          act_next, na_u8,
                                          out_ac + (size_t)t * C_,
                                          out_cells + (size_t)t * N_,
                                          out_anom + t, alpha);
    k_perm_update<<<NUM_ACTIVE_, 256, 0, stream>>>(perm, aidx, x, in_pm);
    k_presyn     <<<32, 256, 0, stream>>>(seg_u8, na_u8, pred_next,
                                          out_pred + (size_t)t * N_);
    k_seg_update <<<N_, 256, 0, stream>>>(seg_f, seg_u8, act_prev, act_next);
  }
}
